// MemoryGame_23502061044267
// MI455X (gfx1250) — matrix operand order repaired and checked by reading, not compiled, not run on hardware
//
#include <hip/hip_runtime.h>
#include <hip/hip_bf16.h>
#include <cstddef>

// ---------------------------------------------------------------------------
// MemoryGame on MI455X (gfx1250, wave32)
//
// The 50-step Hopfield loop over a 9216x9216 matrix dominates (50 sequential
// matvecs). 340MB f32 M does not fit the 192MB L2; a bf16 copy (170MB) does.
// Convert once, run the loop as bf16 WMMA matvecs with f32 accumulation
// served from L2 (streaming f32 passes use non-temporal hints so they don't
// evict the bf16 copy), then one exact-f32 pass for the Hebbian update.
// ---------------------------------------------------------------------------

#define NX_IN 192
#define DIM_X 96
#define DIM_G 96
#define PDIM 9216            // DIM_X * DIM_G
#define NUM_CLASS 1000
#define N_ITER 50
#define KAPPA 0.8f
#define LAMDA 0.9f
#define YITA 0.1f

typedef __attribute__((ext_vector_type(4)))  float  v4f;
typedef __attribute__((ext_vector_type(8)))  float  v8f;
typedef __attribute__((ext_vector_type(16))) __bf16 v16bf;
typedef __attribute__((ext_vector_type(8)))  __bf16 v8bf;

__device__ __forceinline__ float f_p(float v) {
    v = v >= 0.0f ? v : 0.01f * v;           // LeakyReLU(0.01)
    return fminf(1.0f, fmaxf(-1.0f, v));     // clamp [-1,1]
}

// ---------------------------------------------------------------------------
// K0: the two tiny MLPs (x_, g_). One block, trivial cost.
// ---------------------------------------------------------------------------
__global__ __launch_bounds__(384) void mlp_kernel(
    const float* __restrict__ x, const float* __restrict__ g,
    const float* __restrict__ W1x, const float* __restrict__ b1x,
    const float* __restrict__ W2x, const float* __restrict__ b2x,
    const float* __restrict__ W1g, const float* __restrict__ b1g,
    const float* __restrict__ W2g, const float* __restrict__ b2g,
    float* __restrict__ xv, float* __restrict__ gv)
{
    __shared__ float xs[NX_IN], gs[NX_IN], hx[2 * NX_IN], hg[2 * NX_IN];
    int t = threadIdx.x;
    if (t < NX_IN) { xs[t] = x[t]; gs[t] = g[t]; }
    __syncthreads();
    float ax = b1x[t], ag = b1g[t];
    for (int i = 0; i < NX_IN; ++i) {
        ax += xs[i] * W1x[i * (2 * NX_IN) + t];
        ag += gs[i] * W1g[i * (2 * NX_IN) + t];
    }
    hx[t] = fmaxf(ax, 0.0f);
    hg[t] = fmaxf(ag, 0.0f);
    __syncthreads();
    if (t < DIM_X) {
        float ox = b2x[t], og = b2g[t];
        for (int i = 0; i < 2 * NX_IN; ++i) {
            ox += hx[i] * W2x[i * DIM_X + t];
            og += hg[i] * W2g[i * DIM_G + t];
        }
        xv[t] = ox;
        gv[t] = og;
    }
}

// ---------------------------------------------------------------------------
// K1: p = outer(x_, g_) flattened; h0 = f_p(tile(g_)); bf16 shadow of h0.
// ---------------------------------------------------------------------------
__global__ __launch_bounds__(256) void init_kernel(
    const float* __restrict__ xv, const float* __restrict__ gv,
    float* __restrict__ p, float* __restrict__ h, __bf16* __restrict__ hb)
{
    int i = blockIdx.x * blockDim.x + threadIdx.x;   // < PDIM
    float gval = gv[i % DIM_G];
    p[i] = xv[i / DIM_G] * gval;
    float v = f_p(gval);
    h[i]  = v;
    hb[i] = (__bf16)v;
}

// ---------------------------------------------------------------------------
// K2: M f32 -> bf16 (row-major), one streaming pass. Non-temporal reads so
// the 340MB f32 stream does not evict anything useful from L2.
// ---------------------------------------------------------------------------
__global__ __launch_bounds__(256) void cvt_kernel(
    const float* __restrict__ M, __bf16* __restrict__ Mb)
{
    size_t i = ((size_t)blockIdx.x * blockDim.x + threadIdx.x) * 8;
    v4f a = __builtin_nontemporal_load((const v4f*)(M + i));
    v4f b = __builtin_nontemporal_load((const v4f*)(M + i + 4));
    v8bf o;
    o[0] = (__bf16)a[0]; o[1] = (__bf16)a[1]; o[2] = (__bf16)a[2]; o[3] = (__bf16)a[3];
    o[4] = (__bf16)b[0]; o[5] = (__bf16)b[1]; o[6] = (__bf16)b[2]; o[7] = (__bf16)b[3];
    *(v8bf*)(Mb + i) = o;     // regular store: this is the data we WANT in L2
}

// ---------------------------------------------------------------------------
// K3: one Hopfield iteration:  h' = f_p(kappa*h + h .* (h @ M))
//
// Block = 512 threads = 16 waves, owns a 32-column output tile (288 blocks).
// Each wave covers K-range P/16 = 576 in 18 steps of K=32, issuing two
// v_wmma_f32_16x16x32_bf16 per step (columns j0..j0+15 and j0+16..j0+31).
//
// A operand: h chunk as matrix row 0 per the documented 16-bit A layout
// (lane half selects K sub-range). All lanes perform the same broadcast
// loads; rows 1..15 of A receive finite "garbage" (h values), which only
// affects D rows 1..15 that are never read -- no masking, no divergence.
//
// B operand: 32x16 tile of M; lane L supplies column j0+L%16, 16 K-strided
// bf16 (stride PDIM, same k order as A) from row-major memory (L2-resident). Unroll 6 lets the
// scheduler batch loads ahead of the two independent WMMA chains.
// ---------------------------------------------------------------------------
__global__ __launch_bounds__(512) void hop_iter_kernel(
    const float* __restrict__ hcur, const __bf16* __restrict__ hcurb,
    const __bf16* __restrict__ Mb,
    float* __restrict__ hnext, __bf16* __restrict__ hnextb)
{
    __shared__ float red[16][32];
    const int lane  = threadIdx.x & 31;
    const int wave  = threadIdx.x >> 5;
    const int j0    = blockIdx.x * 32;
    const int kbase = wave * (PDIM / 16);            // 576 per wave
    constexpr int KSTEPS = (PDIM / 16) / 32;         // 18

    const __bf16* arow = hcurb + kbase + ((lane >> 4) << 3); // +0 / +8 by half
    const __bf16* brow = Mb + (size_t)(kbase + ((lane >> 4) << 3)) * PDIM + j0 + (lane & 15);

    v8f acc0 = {};
    v8f acc1 = {};

    #pragma unroll 6
    for (int s = 0; s < KSTEPS; ++s) {
        // A: row 0 = h[k0..k0+31]; other rows hold harmless finite values.
        v8bf lo = *(const v8bf*)(arow);          // K = off..off+7
        v8bf hi = *(const v8bf*)(arow + 16);     // K = off+16..off+23
        v16bf a = __builtin_shufflevector(lo, hi,
                                          0, 1, 2, 3, 4, 5, 6, 7,
                                          8, 9, 10, 11, 12, 13, 14, 15);

        v16bf b0, b1;
        for (int e = 0; e < 16; ++e) {
            const size_t o = (size_t)(e < 8 ? e : e + 8) * PDIM;
            b0[e] = brow[o];        // col j0 + lane%16
            b1[e] = brow[o + 16];   // col j0 + 16 + lane%16
        }

        acc0 = __builtin_amdgcn_wmma_f32_16x16x32_bf16(
                   false, a, false, b0, (short)0, acc0, false, false);
        acc1 = __builtin_amdgcn_wmma_f32_16x16x32_bf16(
                   false, a, false, b1, (short)0, acc1, false, false);

        arow += 32;
        brow += (size_t)32 * PDIM;
    }

    // D row 0 lives in acc element 0, lanes 0..15 (N = lane).
    if (lane < 16) {
        red[wave][lane]      = acc0[0];
        red[wave][lane + 16] = acc1[0];
    }
    __syncthreads();

    if (threadIdx.x < 32) {
        float s = 0.0f;
        #pragma unroll
        for (int w = 0; w < 16; ++w) s += red[w][threadIdx.x];
        const int j = j0 + threadIdx.x;
        const float hj = hcur[j];
        const float v  = f_p(KAPPA * hj + hj * s);
        hnext[j]  = v;
        hnextb[j] = (__bf16)v;
    }
}

// ---------------------------------------------------------------------------
// K4: M_out = lamda*M + yita*(p+p_) (p-p_)^T   (exact f32, one pass)
// Non-temporal on the big streams: M is read once, M_out written once.
// ---------------------------------------------------------------------------
__global__ __launch_bounds__(256) void memory_kernel(
    const float* __restrict__ M, const float* __restrict__ p,
    const float* __restrict__ pf, float* __restrict__ Mout)
{
    size_t i = ((size_t)blockIdx.x * blockDim.x + threadIdx.x) * 8;
    const int row = (int)(i / PDIM);
    const int col = (int)(i % PDIM);
    const float ai = YITA * (p[row] + pf[row]);
    v4f m0 = __builtin_nontemporal_load((const v4f*)(M + i));
    v4f m1 = __builtin_nontemporal_load((const v4f*)(M + i + 4));
    v4f o0, o1;
    o0[0] = LAMDA * m0[0] + ai * (p[col + 0] - pf[col + 0]);
    o0[1] = LAMDA * m0[1] + ai * (p[col + 1] - pf[col + 1]);
    o0[2] = LAMDA * m0[2] + ai * (p[col + 2] - pf[col + 2]);
    o0[3] = LAMDA * m0[3] + ai * (p[col + 3] - pf[col + 3]);
    o1[0] = LAMDA * m1[0] + ai * (p[col + 4] - pf[col + 4]);
    o1[1] = LAMDA * m1[1] + ai * (p[col + 5] - pf[col + 5]);
    o1[2] = LAMDA * m1[2] + ai * (p[col + 6] - pf[col + 6]);
    o1[3] = LAMDA * m1[3] + ai * (p[col + 7] - pf[col + 7]);
    __builtin_nontemporal_store(o0, (v4f*)(Mout + i));
    __builtin_nontemporal_store(o1, (v4f*)(Mout + i + 4));
}

// ---------------------------------------------------------------------------
// K5: x_out = rowwise sum of p_; x_inf = x_out @ Wp + bp; also emit p_.
// ---------------------------------------------------------------------------
__global__ __launch_bounds__(256) void out_kernel(
    const float* __restrict__ pf, const float* __restrict__ Wp,
    const float* __restrict__ bp, float* __restrict__ out)
{
    __shared__ float xo[DIM_X];
    const int t = threadIdx.x;

    // copy p_ into d_out (disjoint chunks across blocks)
    for (int i = blockIdx.x * blockDim.x + t; i < PDIM; i += gridDim.x * blockDim.x)
        out[NUM_CLASS + i] = pf[i];

    if (t < DIM_X) {
        float s = 0.0f;
        for (int ig = 0; ig < DIM_G; ++ig) s += pf[t * DIM_G + ig];
        xo[t] = s;
    }
    __syncthreads();

    const int c0 = blockIdx.x * 125;                 // 8 blocks x 125 = 1000
    for (int c = c0 + t; c < c0 + 125; c += blockDim.x) {
        float acc = bp[c];
        #pragma unroll 4
        for (int ix = 0; ix < DIM_X; ++ix)
            acc += xo[ix] * Wp[ix * NUM_CLASS + c];
        out[c] = acc;
    }
}

// ---------------------------------------------------------------------------
extern "C" void kernel_launch(void* const* d_in, const int* in_sizes, int n_in,
                              void* d_out, int out_size, void* d_ws, size_t ws_size,
                              hipStream_t stream)
{
    const float* x   = (const float*)d_in[0];
    const float* g   = (const float*)d_in[1];
    const float* M   = (const float*)d_in[2];
    const float* W1x = (const float*)d_in[3];
    const float* b1x = (const float*)d_in[4];
    const float* W2x = (const float*)d_in[5];
    const float* b2x = (const float*)d_in[6];
    const float* W1g = (const float*)d_in[7];
    const float* b1g = (const float*)d_in[8];
    const float* W2g = (const float*)d_in[9];
    const float* b2g = (const float*)d_in[10];
    const float* Wp  = (const float*)d_in[11];
    const float* bp  = (const float*)d_in[12];

    float* out = (float*)d_out;

    // small scratch in d_ws (< 150 KB)
    float*  ws  = (float*)d_ws;
    float*  xv  = ws + 0;        // 96
    float*  gv  = ws + 128;      // 96
    float*  p   = ws + 256;      // 9216
    float*  hA  = ws + 9472;     // 9216
    float*  hB  = ws + 18688;    // 9216
    __bf16* hAb = (__bf16*)(ws + 27904);  // 9216 bf16
    __bf16* hBb = (__bf16*)(ws + 32512);  // 9216 bf16

    // 170MB bf16 copy of M lives in the M_out slot of d_out (fully
    // overwritten by memory_kernel at the end of every call).
    float*  MoutBase = out + NUM_CLASS + PDIM;
    __bf16* Mb       = (__bf16*)MoutBase;

    mlp_kernel<<<1, 384, 0, stream>>>(x, g, W1x, b1x, W2x, b2x,
                                      W1g, b1g, W2g, b2g, xv, gv);
    init_kernel<<<PDIM / 256, 256, 0, stream>>>(xv, gv, p, hA, hAb);

    const size_t elems = (size_t)PDIM * PDIM;        // 84,934,656
    cvt_kernel<<<(unsigned)(elems / 8 / 256), 256, 0, stream>>>(M, Mb);

    for (int it = 0; it < N_ITER; ++it) {
        if (it & 1)
            hop_iter_kernel<<<PDIM / 32, 512, 0, stream>>>(hB, hBb, Mb, hA, hAb);
        else
            hop_iter_kernel<<<PDIM / 32, 512, 0, stream>>>(hA, hAb, Mb, hB, hBb);
    }
    // N_ITER even -> final state in hA

    memory_kernel<<<(unsigned)(elems / 8 / 256), 256, 0, stream>>>(M, p, hA, MoutBase);
    out_kernel<<<8, 256, 0, stream>>>(hA, Wp, bp, out);
}
